// GraphAttentionLayer_53558242181311
// MI455X (gfx1250) — matrix operand order repaired and checked by reading, not compiled, not run on hardware
//
#include <hip/hip_runtime.h>
#include <hip/hip_bf16.h>

typedef __attribute__((ext_vector_type(16))) _Float16 v16h;
typedef __attribute__((ext_vector_type(8)))  float    v8f;

#define BPB   16            // graph samples (batches) per block
#define ROWS  (BPB * 12)    // 192 rows of Wh per block
#define FDIM  256
#define SAST  264           // LDS row stride in halves (8-half pad vs bank conflicts)
#define ALPHA 0.2f
#define NEGINF -9e15f

// LDS layout (bytes)
#define OFF_SA   0
#define OFF_SWH  (ROWS * SAST * 2)            // after sA (f16 h tile)
#define OFF_WH1  (OFF_SWH + ROWS * SAST * 2)  // after sWh (f16 Wh tile)
#define OFF_WH2  (OFF_WH1 + ROWS * 4)
#define OFF_ATT  (OFF_WH2 + ROWS * 4)
#define SMEM_BYTES (OFF_ATT + BPB * 16 * 16 * 2)  // ~212.5 KB < 320 KB/WGP

__global__ __launch_bounds__(256, 1)
void gat_fused_kernel(const float* __restrict__ h, const float* __restrict__ W,
                      const float* __restrict__ a, const float* __restrict__ adj,
                      const float* __restrict__ bias, float* __restrict__ out)
{
    extern __shared__ char smem[];
    _Float16* sA   = (_Float16*)(smem + OFF_SA);
    _Float16* sWh  = (_Float16*)(smem + OFF_SWH);
    float*    sWh1 = (float*)   (smem + OFF_WH1);
    float*    sWh2 = (float*)   (smem + OFF_WH2);
    _Float16* satt = (_Float16*)(smem + OFF_ATT);

    const int tid     = threadIdx.x;
    const int lane    = tid & 31;
    const int w       = tid >> 5;           // wave id (8 waves of 32)
    const int halfsel = (lane < 16) ? 0 : 1;
    const int lc      = lane & 15;

    // ---------------- Stage 0: stage h tile into LDS as f16 ----------------
    {
        const float4* hblk = (const float4*)(h + (size_t)blockIdx.x * ROWS * FDIM);
        for (int i = tid; i < ROWS * FDIM / 4; i += 256) {
            float4 v = hblk[i];
            int base = i * 4;
            int row  = base >> 8;     // /256
            int col  = base & 255;
            _Float16* p = sA + row * SAST + col;
            p[0] = (_Float16)v.x; p[1] = (_Float16)v.y;
            p[2] = (_Float16)v.z; p[3] = (_Float16)v.w;
        }
    }
    __syncthreads();

    // ---------------- Stage 1: Wh = h @ W  (f16 WMMA, f32 acc) ----------------
    // Output tiles: 12 (M) x 16 (N). Wave w handles n-tiles {w, w+8}, all 12 m-tiles.
    #pragma unroll 1
    for (int p = 0; p < 2; ++p) {
        const int nt  = w + p * 8;
        const int col = nt * 16 + lc;
        // Preload B fragments for all 8 K-steps (W is L2-resident; cvt f32->f16).
        // B 32x16 f16 layout: lanes 0-15 kb=0, lanes 16-31 kb=8;
        // VGPR j<4: K = kb+2j; VGPR j>=4: K = 16+kb+2(j-4).
        v16h bfrag[8];
        const int kbB = halfsel * 8;
        #pragma unroll
        for (int ks = 0; ks < 8; ++ks) {
            #pragma unroll
            for (int j = 0; j < 8; ++j) {
                int k0 = ks * 32 + ((j & 4) ? 16 : 0) + kbB + 2 * (j & 3);
                bfrag[ks][2*j]   = (_Float16)W[(size_t)k0       * FDIM + col];
                bfrag[ks][2*j+1] = (_Float16)W[(size_t)(k0 + 1) * FDIM + col];
            }
        }
        // A 16x32 f16 layout: lanes 0-15 kb=0, lanes 16-31 kb=8;
        // VGPR j<4: K = kb+2j; VGPR j>=4: K = 16+kb+2(j-4).
        const int kbA = halfsel * 8;
        #pragma unroll 1
        for (int mt = 0; mt < 12; ++mt) {
            const int arow = mt * 16 + lc;
            v8f acc = {};
            #pragma unroll
            for (int ks = 0; ks < 8; ++ks) {
                v16h af;
                #pragma unroll
                for (int j = 0; j < 8; ++j) {
                    int k0 = ks * 32 + ((j & 4) ? 16 : 0) + kbA + 2 * (j & 3);
                    af[2*j]   = sA[arow * SAST + k0];
                    af[2*j+1] = sA[arow * SAST + k0 + 1];
                }
                acc = __builtin_amdgcn_wmma_f32_16x16x32_f16(
                          false, af, false, bfrag[ks], (short)0, acc, false, false);
            }
            // D 16x16 f32: VGPR r -> row r (lanes 0-15) / r+8 (lanes 16-31), col = lane&15
            #pragma unroll
            for (int r = 0; r < 8; ++r) {
                int rl = halfsel ? (r + 8) : r;
                sWh[(mt * 16 + rl) * SAST + nt * 16 + lc] = (_Float16)acc[r];
            }
        }
    }
    __syncthreads();

    // ---------------- Stage 1.5: Wh1 = Wh@a1, Wh2 = Wh@a2 ----------------
    if (tid < ROWS) {
        float s1 = 0.f, s2 = 0.f;
        for (int c = 0; c < FDIM; ++c) {
            float v = (float)sWh[tid * SAST + c];
            s1 += v * a[c];
            s2 += v * a[FDIM + c];
        }
        sWh1[tid] = s1;
        sWh2[tid] = s2;
    }
    __syncthreads();

    // ---------------- Stage 1.75: e -> LeakyReLU -> mask -> softmax ----------------
    if (tid < ROWS) {
        int b = tid / 12, i = tid % 12;
        float att[12];
        float m = -3.4e38f;
        for (int j = 0; j < 12; ++j) {
            float e  = sWh1[b * 12 + i] + sWh2[b * 12 + j];
            e = (e > 0.f) ? e : ALPHA * e;
            float ad = adj[i * 12 + j];
            float v  = (ad > 0.f) ? ad * e : NEGINF;
            att[j] = v;
            m = (v > m) ? v : m;
        }
        float s = 0.f;
        for (int j = 0; j < 12; ++j) { att[j] = __expf(att[j] - m); s += att[j]; }
        float inv = 1.f / s;
        for (int j = 0;  j < 12; ++j) satt[b * 256 + i * 16 + j] = (_Float16)(att[j] * inv);
        for (int j = 12; j < 16; ++j) satt[b * 256 + i * 16 + j] = (_Float16)0.f;
    } else {
        int t2 = tid - ROWS;                 // 0..63 -> zero-pad rows 12..15
        int b = t2 >> 2, i = 12 + (t2 & 3);
        for (int j = 0; j < 16; ++j) satt[b * 256 + i * 16 + j] = (_Float16)0.f;
    }
    __syncthreads();

    // ---------------- Stage 2: h' = att @ Wh + bias, ELU (WMMA, K zero-padded) ----------------
    #pragma unroll 1
    for (int p = 0; p < 2; ++p) {
        const int b = w * 2 + p;                             // local batch 0..15
        const size_t rowbase = ((size_t)blockIdx.x * BPB + b) * 12;
        // A fragment: att (16x16 padded, K 16..31 zero)
        v16h af;
        const int kbA = halfsel * 8;
        #pragma unroll
        for (int j = 0; j < 8; ++j) {
            int k0 = ((j & 4) ? 16 : 0) + kbA + 2 * (j & 3);
            _Float16 v0 = (_Float16)0.f, v1 = (_Float16)0.f;
            if (j < 4) {                                     // k0 < 16 exactly when j < 4
                v0 = satt[b * 256 + lc * 16 + k0];
                v1 = satt[b * 256 + lc * 16 + k0 + 1];
            }
            af[2*j] = v0; af[2*j+1] = v1;
        }
        const int kbB = halfsel * 8;
        #pragma unroll 1
        for (int nt = 0; nt < 16; ++nt) {
            v16h bf;
            #pragma unroll
            for (int j = 0; j < 8; ++j) {
                int k = ((j & 4) ? 16 : 0) + kbB + 2 * (j & 3); // node index, pad >=12 with 0
                bf[2*j]   = (k     < 12) ? sWh[(b * 12 + k    ) * SAST + nt * 16 + lc] : (_Float16)0.f;
                bf[2*j+1] = (k + 1 < 12) ? sWh[(b * 12 + k + 1) * SAST + nt * 16 + lc] : (_Float16)0.f;
            }
            v8f c = {};
            c = __builtin_amdgcn_wmma_f32_16x16x32_f16(
                    false, af, false, bf, (short)0, c, false, false);
            int   colg = nt * 16 + lc;
            float bv   = bias[colg];
            #pragma unroll
            for (int r = 0; r < 8; ++r) {
                int il = halfsel ? (r + 8) : r;
                float v = c[r] + bv;
                v = (v > 0.f) ? v : (__expf(v) - 1.f);       // ELU(alpha=1)
                if (il < 12) out[(rowbase + il) * FDIM + colg] = v;
            }
        }
    }
}

extern "C" void kernel_launch(void* const* d_in, const int* in_sizes, int n_in,
                              void* d_out, int out_size, void* d_ws, size_t ws_size,
                              hipStream_t stream) {
    (void)in_sizes; (void)n_in; (void)d_ws; (void)ws_size; (void)out_size;
    const float* h    = (const float*)d_in[0];
    const float* W    = (const float*)d_in[1];
    const float* a    = (const float*)d_in[2];
    const float* adj  = (const float*)d_in[3];
    const float* bias = (const float*)d_in[4];
    float* out = (float*)d_out;

    const int nblocks = (16384 * 12) / ROWS;   // 1024
    gat_fused_kernel<<<nblocks, 256, SMEM_BYTES, stream>>>(h, W, a, adj, bias, out);
}
